// EmbedMatcher_LSTMAE_17635135717981
// MI455X (gfx1250) — compile-verified, matrix operand order checked by reading, not run on hardware
//
#include <hip/hip_runtime.h>

// ---------------------------------------------------------------------------
// CDNA5 (gfx1250) wave32 WMMA implementation of the EmbedMatcher reference.
// Big GEMMs use v_wmma_f32_16x16x32_bf16 (bf16 in, f32 accumulate).
// ---------------------------------------------------------------------------

typedef __bf16 bf16_t;
typedef __attribute__((ext_vector_type(16))) __bf16 v16bf;
typedef __attribute__((ext_vector_type(8)))  __bf16 v8bf;
typedef __attribute__((ext_vector_type(8)))  float  v8f;

__device__ __forceinline__ bf16_t f2bf(float f) {
  union { float f; unsigned u; } v; v.f = f;
  unsigned r = v.u + 0x7FFFu + ((v.u >> 16) & 1u);   // round-to-nearest-even
  unsigned short h = (unsigned short)(r >> 16);
  return __builtin_bit_cast(bf16_t, h);
}

__device__ __forceinline__ float sigm(float x) { return 1.f / (1.f + __expf(-x)); }

// Load a 16x32 bf16 WMMA fragment from row-major [rows][ldk] storage.
// ISA 7.12.2: lane L -> row L&15, K chunks {kb..kb+7, kb+16..kb+23}, kb=(L&16)?8:0.
// Identical pattern serves A ([M,K]) and B (= weight stored [N,K] row-major).
__device__ __forceinline__ v16bf load_frag(const bf16_t* p, int ldk, int lane) {
  const int r  = lane & 15;
  const int kb = (lane & 16) ? 8 : 0;
  const bf16_t* q = p + (long)r * ldk + kb;
  v8bf lo = *(const v8bf*)(q);
  v8bf hi = *(const v8bf*)(q + 16);
  v16bf out;
#pragma unroll
  for (int i = 0; i < 8; ++i) { out[i] = lo[i]; out[8 + i] = hi[i]; }
  return out;
}

__device__ __forceinline__ v8f wmma_bf16(v16bf a, v16bf b, v8f c) {
  // D = A*B + C, f32 accumulate  ->  v_wmma_f32_16x16x32_bf16
  return __builtin_amdgcn_wmma_f32_16x16x32_bf16(false, a, false, b, (short)0, c,
                                                 false, false);
}

// ---------------------------------------------------------------------------
// Weight prep: f32 -> bf16
// ---------------------------------------------------------------------------
__global__ void conv_bf16_kernel(const float* __restrict__ s, bf16_t* __restrict__ d, int n) {
  int i = blockIdx.x * 256 + threadIdx.x;
  if (i < n) d[i] = f2bf(s[i]);
}

// Build Bcat[4096][1536] = [qWih | qWhh] (bf16) and qbias = qBih + qBhh.
__global__ void bcat_kernel(const float* __restrict__ qWih, const float* __restrict__ qWhh,
                            bf16_t* __restrict__ Bcat,
                            const float* __restrict__ qBih, const float* __restrict__ qBhh,
                            float* __restrict__ qbias) {
  long idx = (long)blockIdx.x * 256 + threadIdx.x;       // 4096*1536 total
  int nrow = (int)(idx / 1536), col = (int)(idx % 1536);
  float v = (col < 512) ? qWih[(long)nrow * 512 + col]
                        : qWhh[(long)nrow * 1024 + (col - 512)];
  Bcat[idx] = f2bf(v);
  if (idx < 4096) qbias[idx] = qBih[idx] + qBhh[idx];
}

// ---------------------------------------------------------------------------
// Neighbor encoder: one row (30 neighbors) per workgroup.
// cat[30,512] @ naW^T -> tanh -> score dot nuW -> softmax(30) -> weighted ent -> tanh
// ---------------------------------------------------------------------------
__global__ __launch_bounds__(256) void neighbor_kernel(
    const int* __restrict__ conn,          // [N][30][2]
    const float* __restrict__ emb,         // [V+1][256]
    const bf16_t* __restrict__ naWb,       // [256][512] bf16
    const float* __restrict__ naB,         // [256]
    const float* __restrict__ nuW,         // [256]
    float* __restrict__ outF,              // [N][512] f32
    bf16_t* __restrict__ outB,             // [N][512] bf16
    int colofs)
{
  __shared__ bf16_t catA[32][512];         // [rel|ent] rows, padded to 32
  __shared__ float  entF[30][256];
  __shared__ float  score[32];
  __shared__ float  att[32];
  const int n   = blockIdx.x;
  const int tid = threadIdx.x;

  for (int idx = tid; idx < 30 * 512; idx += 256) {
    int k = idx >> 9, j = idx & 511;
    int which = j >> 8, e = j & 255;
    int row = conn[((long)n * 30 + k) * 2 + which];
    float v = emb[(long)row * 256 + e];
    catA[k][j] = f2bf(v);
    if (which) entF[k][e] = v;
  }
  for (int idx = tid; idx < 2 * 512; idx += 256)
    catA[30 + (idx >> 9)][idx & 511] = f2bf(0.f);
  if (tid < 32) score[tid] = 0.f;
  __syncthreads();

  const int wave = tid >> 5, lane = tid & 31;
#pragma unroll
  for (int t = 0; t < 4; ++t) {
    const int tile = wave + t * 8;         // 32 tiles: 2 M-tiles x 16 N-tiles
    const int mt = tile >> 4, nt = tile & 15;
    v8f acc = {0.f, 0.f, 0.f, 0.f, 0.f, 0.f, 0.f, 0.f};
    for (int kk = 0; kk < 16; ++kk) {
      v16bf a = load_frag(&catA[mt * 16][kk * 32], 512, lane);
      v16bf b = load_frag(naWb + (long)(nt * 16) * 512 + kk * 32, 512, lane);
      acc = wmma_bf16(a, b, acc);
    }
    const int   ncol = nt * 16 + (lane & 15);
    const float bias = naB[ncol];
    const float nu   = nuW[ncol];
#pragma unroll
    for (int j = 0; j < 8; ++j) {
      float val = tanhf(acc[j] + bias) * nu;
      val += __shfl_xor(val, 1, 32);
      val += __shfl_xor(val, 2, 32);
      val += __shfl_xor(val, 4, 32);
      val += __shfl_xor(val, 8, 32);
      if ((lane & 15) == 0) {
        int m = mt * 16 + ((lane >> 4) << 3) + j;
        atomicAdd(&score[m], val);
      }
    }
  }
  __syncthreads();
  if (tid == 0) {                           // softmax over 30 real neighbors
    float mx = -3.0e38f;
    for (int k = 0; k < 30; ++k) mx = fmaxf(mx, score[k]);
    float s = 0.f;
    for (int k = 0; k < 30; ++k) { float e = __expf(score[k] - mx); att[k] = e; s += e; }
    float inv = 1.f / s;
    for (int k = 0; k < 30; ++k) att[k] *= inv;
  }
  __syncthreads();
  {
    float s = 0.f;
    for (int k = 0; k < 30; ++k) s += att[k] * entF[k][tid];
    float r = tanhf(s);
    outF[(long)n * 512 + colofs + tid] = r;
    outB[(long)n * 512 + colofs + tid] = f2bf(r);
  }
}

// ---------------------------------------------------------------------------
// Generic bf16 WMMA GEMM: C[M,N] = A[M,K] @ B[N,K]^T (+bias / +resid / relu)
// Block = 256 thr (8 waves), 128x128 C tile; wave w owns N-strip w, 8 M-subtiles.
// mode 0: relu(acc+bias) -> bf16    mode 1: acc+bias+resid -> f32    mode 2: acc+bias -> f32
// M,N multiples of 128; K multiple of 32.
// ---------------------------------------------------------------------------
__global__ __launch_bounds__(256) void gemm_bf16_kernel(
    const bf16_t* __restrict__ A, const bf16_t* __restrict__ B,
    int M, int N, int K,
    const float* __restrict__ bias, int mode,
    float* __restrict__ Cf, bf16_t* __restrict__ Cb,
    const float* __restrict__ resid)
{
  __shared__ bf16_t Atile[128][32];
  const int mBase = blockIdx.y * 128;
  const int nBase = blockIdx.x * 128;
  const int tid  = threadIdx.x;
  const int wave = tid >> 5, lane = tid & 31;

  v8f acc[8];
#pragma unroll
  for (int i = 0; i < 8; ++i) acc[i] = (v8f){0.f, 0.f, 0.f, 0.f, 0.f, 0.f, 0.f, 0.f};

  const bf16_t* Bw = B + (long)(nBase + wave * 16) * K;

  for (int k0 = 0; k0 < K; k0 += 32) {
    __syncthreads();                       // guard previous Atile readers
    {
      int row = tid >> 1, hf = tid & 1;    // 2 threads per row, 16 elems each
      const v8bf* src = (const v8bf*)(A + (long)(mBase + row) * K + k0 + hf * 16);
      v8bf x0 = src[0], x1 = src[1];
      *(v8bf*)&Atile[row][hf * 16]     = x0;
      *(v8bf*)&Atile[row][hf * 16 + 8] = x1;
    }
    __syncthreads();
    v16bf bfrag = load_frag(Bw + k0, K, lane);
#pragma unroll
    for (int ms = 0; ms < 8; ++ms) {
      v16bf afrag = load_frag(&Atile[ms * 16][0], 32, lane);
      acc[ms] = wmma_bf16(afrag, bfrag, acc[ms]);
    }
  }

  const int   n  = nBase + wave * 16 + (lane & 15);
  const float bv = bias ? bias[n] : 0.f;
#pragma unroll
  for (int ms = 0; ms < 8; ++ms) {
#pragma unroll
    for (int j = 0; j < 8; ++j) {
      int  m = mBase + ms * 16 + ((lane >> 4) << 3) + j;
      float v = acc[ms][j] + bv;
      long  o = (long)m * N + n;
      if (mode == 0)      Cb[o] = f2bf(fmaxf(v, 0.f));
      else if (mode == 1) Cf[o] = v + resid[o];
      else                Cf[o] = v;
    }
  }
}

// ---------------------------------------------------------------------------
// Row LayerNorm (torch semantics: unbiased std, eps on sigma). Row = 512.
// Writes query_g f32 (to d_out region) and bf16 copy into Acat[:,0:512].
// ---------------------------------------------------------------------------
__global__ __launch_bounds__(256) void ln_kernel(const float* __restrict__ X,
                                                 const float* __restrict__ lnA,
                                                 const float* __restrict__ lnB,
                                                 float* __restrict__ qgF,
                                                 bf16_t* __restrict__ Acat)
{
  __shared__ float red[256];
  const int row = blockIdx.x, tid = threadIdx.x;
  float x0 = X[(long)row * 512 + tid];
  float x1 = X[(long)row * 512 + 256 + tid];
  red[tid] = x0 + x1; __syncthreads();
  for (int s = 128; s > 0; s >>= 1) { if (tid < s) red[tid] += red[tid + s]; __syncthreads(); }
  float mu = red[0] / 512.f; __syncthreads();
  float d0 = x0 - mu, d1 = x1 - mu;
  red[tid] = d0 * d0 + d1 * d1; __syncthreads();
  for (int s = 128; s > 0; s >>= 1) { if (tid < s) red[tid] += red[tid + s]; __syncthreads(); }
  float sg  = sqrtf(red[0] / 511.f);
  float inv = 1.f / (sg + 1e-3f);
  float y0 = d0 * inv * lnA[tid]       + lnB[tid];
  float y1 = d1 * inv * lnA[256 + tid] + lnB[256 + tid];
  qgF[(long)row * 512 + tid]        = y0;
  qgF[(long)row * 512 + 256 + tid]  = y1;
  Acat[(long)row * 1536 + tid]       = f2bf(y0);
  Acat[(long)row * 1536 + 256 + tid] = f2bf(y1);
}

// ---------------------------------------------------------------------------
// Support branch (batch-1 sequential): support encoder + LSTM autoencoder +
// attention pooling. Single workgroup, 512 threads, plain f32 VALU.
// ---------------------------------------------------------------------------
__device__ __forceinline__ float block_reduce_sum512(float v, float* red, int tid) {
  red[tid] = v; __syncthreads();
  for (int s = 256; s > 0; s >>= 1) { if (tid < s) red[tid] += red[tid + s]; __syncthreads(); }
  float r = red[0]; __syncthreads();
  return r;
}

__global__ __launch_bounds__(512) void support_branch_kernel(
    const float* __restrict__ snF,
    const float* __restrict__ p1W, const float* __restrict__ p1b,
    const float* __restrict__ p2W, const float* __restrict__ p2b,
    const float* __restrict__ lnA, const float* __restrict__ lnB,
    const float* __restrict__ encWih, const float* __restrict__ encWhh,
    const float* __restrict__ encBih, const float* __restrict__ encBhh,
    const float* __restrict__ decWih, const float* __restrict__ decWhh,
    const float* __restrict__ decBih, const float* __restrict__ decBhh,
    const float* __restrict__ saW, const float* __restrict__ saB,
    const float* __restrict__ suW, const float* __restrict__ suB,
    float* __restrict__ d_out, float* __restrict__ ws_sge)
{
  (void)saW; (void)saB; (void)suW; (void)suB;  // softmax over size-1 axis == 1.0
  const int tid = threadIdx.x;
  __shared__ float sg[3][512];
  __shared__ float eo[3][512];
  __shared__ float dec[3][512];
  __shared__ float h1[1024];
  __shared__ float ov[512];
  __shared__ float h[512], c[512], xv[512];
  __shared__ float g[2048];
  __shared__ float red[512];

  // support encoder (3 rows)
  for (int r = 0; r < 3; ++r) {
    const float* x = snF + (long)r * 512;
    for (int u = tid; u < 1024; u += 512) {
      float s = p1b[u];
      const float* w = p1W + (long)u * 512;
      for (int k = 0; k < 512; ++k) s += w[k] * x[k];
      h1[u] = fmaxf(s, 0.f);
    }
    __syncthreads();
    {
      float s = p2b[tid];
      const float* w = p2W + (long)tid * 1024;
      for (int k = 0; k < 1024; ++k) s += w[k] * h1[k];
      ov[tid] = s + x[tid];
    }
    __syncthreads();
    float mu  = block_reduce_sum512(ov[tid], red, tid) / 512.f;
    float dd  = ov[tid] - mu;
    float var = block_reduce_sum512(dd * dd, red, tid) / 511.f;
    float sig = sqrtf(var);
    sg[r][tid] = dd / (sig + 1e-3f) * lnA[tid] + lnB[tid];
    __syncthreads();
  }

  // encoder LSTM (seq=3, batch=1), gates i,f,g,o
  h[tid] = 0.f; c[tid] = 0.f;
  __syncthreads();
  for (int st = 0; st < 3; ++st) {
    for (int u = tid; u < 2048; u += 512) {
      float s = encBih[u] + encBhh[u];
      const float* wi = encWih + (long)u * 512;
      const float* wh = encWhh + (long)u * 512;
      for (int k = 0; k < 512; ++k) s += wi[k] * sg[st][k];
      for (int k = 0; k < 512; ++k) s += wh[k] * h[k];
      g[u] = s;
    }
    __syncthreads();
    {
      float iv = g[tid], fv = g[512 + tid], gv = g[1024 + tid], oo = g[1536 + tid];
      float cc = sigm(fv) * c[tid] + sigm(iv) * tanhf(gv);
      c[tid] = cc;
      float hh = sigm(oo) * tanhf(cc);
      h[tid] = hh; eo[st][tid] = hh;
    }
    __syncthreads();
  }

  // decoder LSTM: init from encoder final state, x = enc_out[-1], feed back h
  xv[tid] = eo[2][tid];
  __syncthreads();
  for (int st = 0; st < 3; ++st) {
    for (int u = tid; u < 2048; u += 512) {
      float s = decBih[u] + decBhh[u];
      const float* wi = decWih + (long)u * 512;
      const float* wh = decWhh + (long)u * 512;
      for (int k = 0; k < 512; ++k) s += wi[k] * xv[k];
      for (int k = 0; k < 512; ++k) s += wh[k] * h[k];
      g[u] = s;
    }
    __syncthreads();
    {
      float iv = g[tid], fv = g[512 + tid], gv = g[1024 + tid], oo = g[1536 + tid];
      float cc = sigm(fv) * c[tid] + sigm(iv) * tanhf(gv);
      c[tid] = cc;
      float hh = sigm(oo) * tanhf(cc);
      h[tid] = hh; dec[st][tid] = hh;
    }
    __syncthreads();
    xv[tid] = h[tid];
    __syncthreads();
  }

  // ae_loss = mean((support_g - dec)^2)
  float ls = 0.f;
  for (int r = 0; r < 3; ++r) { float dd = sg[r][tid] - dec[r][tid]; ls += dd * dd; }
  float tot = block_reduce_sum512(ls, red, tid);
  if (tid == 0) d_out[2048] = tot / 1536.f;

  // sge = support_g + enc_out; attention weights = softmax over size-1 == 1
  {
    float v = (sg[0][tid] + eo[0][tid]) + (sg[1][tid] + eo[1][tid]) + (sg[2][tid] + eo[2][tid]);
    d_out[2049 + tid] = v;
    ws_sge[tid] = v;
  }
}

// ---------------------------------------------------------------------------
// Query process loop helpers
// ---------------------------------------------------------------------------
__global__ void init_state_kernel(float* __restrict__ c, bf16_t* __restrict__ Acat) {
  long idx = (long)blockIdx.x * 256 + threadIdx.x;   // 2048*1024
  c[idx] = 0.f;
  int row = (int)(idx >> 10), u = (int)(idx & 1023);
  Acat[(long)row * 1536 + 512 + u] = f2bf(0.f);      // h_r init = zeros
}

__global__ void gate_kernel(const float* __restrict__ g, float* __restrict__ c,
                            const float* __restrict__ qg, const float* __restrict__ sge,
                            float* __restrict__ hout, bf16_t* __restrict__ Acat) {
  long idx = (long)blockIdx.x * 256 + threadIdx.x;   // 2048*1024
  int row = (int)(idx >> 10), u = (int)(idx & 1023);
  const float* gr = g + (long)row * 4096;            // gates i,f,g,o (1024 each)
  float iv = gr[u], fv = gr[1024 + u], gv = gr[2048 + u], oo = gr[3072 + u];
  float cc = sigm(fv) * c[idx] + sigm(iv) * tanhf(gv);
  c[idx] = cc;
  float hh = sigm(oo) * tanhf(cc);
  if (u < 512) {
    float hval = qg[(long)row * 512 + u] + hh;       // h = query_g + h_r_[:, :D]
    hout[(long)row * 512 + u] = hval;
    Acat[(long)row * 1536 + 512 + u] = f2bf(hval);
  } else {
    // attn softmax over size-1 axis == 1 -> r = support_g_encoder broadcast
    Acat[(long)row * 1536 + 512 + u] = f2bf(sge[u - 512]);
  }
}

__global__ __launch_bounds__(256) void matching_kernel(const float* __restrict__ h,
                                                       const float* __restrict__ sge,
                                                       float* __restrict__ out) {
  __shared__ float sv[512];
  int tid = threadIdx.x;
  sv[tid] = sge[tid]; sv[256 + tid] = sge[256 + tid];
  __syncthreads();
  int row = blockIdx.x * 256 + tid;
  const float* hr = h + (long)row * 512;
  float acc = 0.f;
  for (int k = 0; k < 512; ++k) acc += hr[k] * sv[k];
  out[row] = acc;
}

// ---------------------------------------------------------------------------
// Host launch
// ---------------------------------------------------------------------------
extern "C" void kernel_launch(void* const* d_in, const int* in_sizes, int n_in,
                              void* d_out, int out_size, void* d_ws, size_t ws_size,
                              hipStream_t stream) {
  (void)in_sizes; (void)n_in; (void)out_size; (void)ws_size;
  // input order per setup_inputs()
  const int*   qlc    = (const int*)  d_in[2];
  const int*   qrc    = (const int*)  d_in[4];
  const int*   slc    = (const int*)  d_in[6];
  const int*   src_   = (const int*)  d_in[8];
  const float* emb    = (const float*)d_in[10];
  const float* naW    = (const float*)d_in[11]; const float* naB  = (const float*)d_in[12];
  const float* nuW    = (const float*)d_in[13];
  const float* p1W    = (const float*)d_in[15]; const float* p1b  = (const float*)d_in[16];
  const float* p2W    = (const float*)d_in[17]; const float* p2b  = (const float*)d_in[18];
  const float* lnA    = (const float*)d_in[19]; const float* lnB  = (const float*)d_in[20];
  const float* encWih = (const float*)d_in[21]; const float* encWhh = (const float*)d_in[22];
  const float* encBih = (const float*)d_in[23]; const float* encBhh = (const float*)d_in[24];
  const float* decWih = (const float*)d_in[25]; const float* decWhh = (const float*)d_in[26];
  const float* decBih = (const float*)d_in[27]; const float* decBhh = (const float*)d_in[28];
  const float* saW    = (const float*)d_in[29]; const float* saB  = (const float*)d_in[30];
  const float* suW    = (const float*)d_in[31]; const float* suB  = (const float*)d_in[32];
  const float* qWih   = (const float*)d_in[33]; const float* qWhh = (const float*)d_in[34];
  const float* qBih   = (const float*)d_in[35]; const float* qBhh = (const float*)d_in[36];

  // workspace carve (256B aligned)
  char* ws = (char*)d_ws; size_t off = 0;
  auto carve = [&](size_t bytes) -> void* {
    void* p = ws + off; off += (bytes + 255) & ~(size_t)255; return p;
  };
  bf16_t* naWb = (bf16_t*)carve((size_t)256 * 512 * 2);
  bf16_t* p1Wb = (bf16_t*)carve((size_t)1024 * 512 * 2);
  bf16_t* p2Wb = (bf16_t*)carve((size_t)512 * 1024 * 2);
  bf16_t* Bcat = (bf16_t*)carve((size_t)4096 * 1536 * 2);
  float*  qbias= (float*) carve((size_t)4096 * 4);
  float*  qnF  = (float*) carve((size_t)2048 * 512 * 4);
  bf16_t* qnB  = (bf16_t*)carve((size_t)2048 * 512 * 2);
  float*  snF  = (float*) carve((size_t)3 * 512 * 4);
  bf16_t* snB  = (bf16_t*)carve((size_t)3 * 512 * 2);
  bf16_t* hB   = (bf16_t*)carve((size_t)2048 * 1024 * 2);
  float*  oRes = (float*) carve((size_t)2048 * 512 * 4);
  bf16_t* Acat = (bf16_t*)carve((size_t)2048 * 1536 * 2);
  float*  gbuf = (float*) carve((size_t)2048 * 4096 * 4);
  float*  cbuf = (float*) carve((size_t)2048 * 1024 * 4);
  float*  hbuf = (float*) carve((size_t)2048 * 512 * 4);
  float*  sgeW = (float*) carve((size_t)512 * 4);

  float* outF = (float*)d_out;
  float* qgF  = outF + 2561;   // query_g output region [2048*512]

  // 1) weight prep
  conv_bf16_kernel<<<(256 * 512 + 255) / 256, 256, 0, stream>>>(naW, naWb, 256 * 512);
  conv_bf16_kernel<<<(1024 * 512 + 255) / 256, 256, 0, stream>>>(p1W, p1Wb, 1024 * 512);
  conv_bf16_kernel<<<(512 * 1024 + 255) / 256, 256, 0, stream>>>(p2W, p2Wb, 512 * 1024);
  bcat_kernel<<<(4096 * 1536) / 256, 256, 0, stream>>>(qWih, qWhh, Bcat, qBih, qBhh, qbias);

  // 2) neighbor encoders (left -> cols 0:256, right -> cols 256:512)
  neighbor_kernel<<<2048, 256, 0, stream>>>(qlc,  emb, naWb, naB, nuW, qnF, qnB, 0);
  neighbor_kernel<<<2048, 256, 0, stream>>>(qrc,  emb, naWb, naB, nuW, qnF, qnB, 256);
  neighbor_kernel<<<3,    256, 0, stream>>>(slc,  emb, naWb, naB, nuW, snF, snB, 0);
  neighbor_kernel<<<3,    256, 0, stream>>>(src_, emb, naWb, naB, nuW, snF, snB, 256);

  // 3) query support-encoder: relu GEMM, residual GEMM, LayerNorm -> query_g
  gemm_bf16_kernel<<<dim3(1024 / 128, 2048 / 128), 256, 0, stream>>>(
      qnB, p1Wb, 2048, 1024, 512, p1b, 0, nullptr, hB, nullptr);
  gemm_bf16_kernel<<<dim3(512 / 128, 2048 / 128), 256, 0, stream>>>(
      hB, p2Wb, 2048, 512, 1024, p2b, 1, oRes, nullptr, qnF);
  ln_kernel<<<2048, 256, 0, stream>>>(oRes, lnA, lnB, qgF, Acat);

  // 4) support branch (encoder + LSTM AE + pooling) -> ae_loss, support_g_encoder
  support_branch_kernel<<<1, 512, 0, stream>>>(
      snF, p1W, p1b, p2W, p2b, lnA, lnB,
      encWih, encWhh, encBih, encBhh, decWih, decWhh, decBih, decBhh,
      saW, saB, suW, suB, outF, sgeW);

  // 5) query process loop: 4 x (fused GEMM [2048,1536]x[4096,1536]^T + gates)
  init_state_kernel<<<(2048 * 1024) / 256, 256, 0, stream>>>(cbuf, Acat);
  for (int s = 0; s < 4; ++s) {
    gemm_bf16_kernel<<<dim3(4096 / 128, 2048 / 128), 256, 0, stream>>>(
        Acat, Bcat, 2048, 4096, 1536, qbias, 2, gbuf, nullptr, nullptr);
    gate_kernel<<<(2048 * 1024) / 256, 256, 0, stream>>>(gbuf, cbuf, qgF, sgeW, hbuf, Acat);
  }

  // 6) matching scores
  matching_kernel<<<2048 / 256, 256, 0, stream>>>(hbuf, sgeW, outF);
}
